// BlockCirculantLinear_1589137900074
// MI455X (gfx1250) — matrix operand order repaired and checked by reading, not compiled, not run on hardware
//
#include <hip/hip_runtime.h>

// ---------------------------------------------------------------------------
// Block-circulant linear == dense GEMM  y = (x*sign) @ W^T + bias
// with W[o*512+t, i*512+s] = g_{o,i}[(t-s) mod 512],
// g = real inverse-DFT of the complex spectrum Lam.
// GEMM is matrix-core bound on MI455X -> bf16 WMMA (f32 accumulate),
// double-buffered LDS, B tile streamed by the Tensor Data Mover.
// ---------------------------------------------------------------------------

typedef __attribute__((ext_vector_type(16))) __bf16        v16bf;
typedef __attribute__((ext_vector_type(8)))  float         v8f;
typedef __attribute__((ext_vector_type(4)))  unsigned int  v4u;
typedef __attribute__((ext_vector_type(4)))  unsigned int  u32x4;
typedef __attribute__((ext_vector_type(8)))  int           i32x8;
typedef __attribute__((ext_vector_type(4)))  int           i32x4;

union FragBF {
    v16bf bf;
    v4u   q[2];
};

#if __has_builtin(__builtin_amdgcn_tensor_load_to_lds) && \
    __has_builtin(__builtin_amdgcn_s_wait_tensorcnt)
#define USE_TDM 1
#else
#define USE_TDM 0
#endif

__device__ __forceinline__ unsigned short f32_to_bf16_rne(float f) {
    unsigned u = __builtin_bit_cast(unsigned, f);
    unsigned r = u + 0x7FFFu + ((u >> 16) & 1u);   // round-to-nearest-even
    return (unsigned short)(r >> 16);
}

// Low 32 bits of a flat pointer into the LDS aperture == LDS byte address
// (ISA 10.2).
__device__ __forceinline__ unsigned lds_byte_addr(const void* p) {
    return (unsigned)(unsigned long long)p;
}

// ---------------------------------------------------------------------------
// Kernel 1: g[(o*8+i)*512 + d] = (1/512) * sum_f Re(Lam) cos(2pi f d/512)
//                                         - Im(Lam) sin(2pi f d/512)
// ---------------------------------------------------------------------------
__global__ void build_g_kernel(const float* __restrict__ re,
                               const float* __restrict__ im,
                               float* __restrict__ g) {
    int idx = blockIdx.x * blockDim.x + threadIdx.x;   // 0 .. 64*512-1
    int blk = idx >> 9;
    int d   = idx & 511;
    float theta = (6.28318530717958647692f / 512.0f) * (float)d;
    float st, ct;
    __sincosf(theta, &st, &ct);
    const float* rp = re + (blk << 9);
    const float* ip = im + (blk << 9);
    float c = 1.0f, s = 0.0f, acc = 0.0f;
    for (int f = 0; f < 512; ++f) {
        acc += rp[f] * c - ip[f] * s;
        float cn = c * ct - s * st;
        s        = s * ct + c * st;
        c        = cn;
    }
    g[idx] = acc * (1.0f / 512.0f);
}

// ---------------------------------------------------------------------------
// Kernel 2: W_bf16[r*4096 + c] = bf16( g_{o,i}[(t-s)&511] * sign_flip[c] )
// ---------------------------------------------------------------------------
__global__ void build_w_kernel(const float* __restrict__ g,
                               const float* __restrict__ sign_flip,
                               unsigned short* __restrict__ W) {
    int idx = blockIdx.x * blockDim.x + threadIdx.x;   // 0 .. 4096*4096-1
    int r = idx >> 12;
    int c = idx & 4095;
    int o = r >> 9, t = r & 511;
    int i = c >> 9, s = c & 511;
    float v = g[((o * 8 + i) << 9) + ((t - s) & 511)] * sign_flip[c];
    W[idx] = f32_to_bf16_rne(v);
}

// ---------------------------------------------------------------------------
// Kernel 3: out[16384 x 4096] = X * W^T + bias, bf16 WMMA.
// 256 threads = 8 wave32, tile 128(M) x 128(N), TILE_K = 32, double-buffered.
// B tile: one TDM tensor_load_to_lds per K-step (wave 0), padded by the TDM
// to an 80-byte row stride for conflict-free ds_load_b128 fragment reads.
// ---------------------------------------------------------------------------
#define TILE_M   128
#define TILE_N   128
#define TILE_K   32
#define LDS_PAD  40     // row stride in ushorts (80 B)

#if USE_TDM
__device__ __forceinline__ void tdm_load_b_tile(const unsigned short* gptr,
                                                unsigned lds_addr) {
    unsigned long long ga = (unsigned long long)gptr;
    u32x4 g0;
    g0.x = 1u;                                               // count=1 (valid)
    g0.y = lds_addr;                                         // lds_addr (bytes)
    g0.z = (unsigned)(ga & 0xFFFFFFFFull);                   // global_addr[31:0]
    g0.w = (unsigned)((ga >> 32) & 0x1FFFFFFull) | (2u << 30); // addr[56:32]|type=2

    i32x8 g1;
    // d0: wg_mask=0 | data_size=1(2B)<<16 | pad_enable<<20 |
    //     pad_interval=3 (16 DWORDs)<<22 | pad_amount=3 (4 DWORDs)<<25
    g1[0] = (1 << 16) | (1 << 20) | (3 << 22) | (3 << 25);
    // d1: atomic_barrier_addr[15:0]=0 | tensor_dim0[15:0]   (tensor_dim0=4096)
    g1[1] = (int)((4096u & 0xFFFFu) << 16);
    // d2: tensor_dim0[31:16] | tensor_dim1[15:0]            (tensor_dim1=4096)
    g1[2] = (int)((4096u >> 16) | ((4096u & 0xFFFFu) << 16));
    // d3: tensor_dim1[31:16] | tile_dim0=32
    g1[3] = (int)((4096u >> 16) | (32u << 16));
    // d4: tile_dim1=128 | tile_dim2=0
    g1[4] = 128;
    // d5: tensor_dim0_stride[31:0] = 4096 elements (8 KB rows)
    g1[5] = 4096;
    g1[6] = 0;      // dim0_stride[47:32] | dim1_stride[15:0]
    g1[7] = 0;      // dim1_stride[47:16]

    i32x4 z4 = {0, 0, 0, 0};
#if __clang_major__ >= 23
    i32x8 z8 = {0, 0, 0, 0, 0, 0, 0, 0};
    __builtin_amdgcn_tensor_load_to_lds(g0, g1, z4, z4, z8, 0);
#else
    __builtin_amdgcn_tensor_load_to_lds(g0, g1, z4, z4, 0);
#endif
}
#endif

__global__ __launch_bounds__(256)
void bcl_gemm_kernel(const float* __restrict__ X,
                     const unsigned short* __restrict__ W,
                     const float* __restrict__ bias,
                     float* __restrict__ out) {
    __shared__ __align__(16) unsigned short Als[2][TILE_M * LDS_PAD];
    __shared__ __align__(16) unsigned short Bls[2][TILE_N * LDS_PAD];

    const int tid   = threadIdx.x;
    const int lane  = tid & 31;
    const int wave  = tid >> 5;
    const int wm    = wave & 3;      // 32-row slab within tile
    const int wn    = wave >> 2;     // 64-col slab within tile

    const int nBase = blockIdx.x * TILE_N;
    const int mBase = blockIdx.y * TILE_M;

    const int frow  = tid >> 1;      // 0..127 : fill row
    const int fhalf = tid & 1;       // 16-element half of a 32-wide row

    const int khalf = lane >> 4;     // K-half select per ISA 16-bit layouts
    const int lm    = lane & 15;

    // A tile fill: f32 -> bf16 (RNE) on the way into LDS; co-executes w/ WMMA.
    auto fillA = [&](int buf, int kb) {
        const float4* s4 =
            (const float4*)(X + (size_t)(mBase + frow) * 4096 + kb + fhalf * 16);
        __builtin_prefetch(s4 + 16, 0, 1);   // 2 K-tiles ahead -> global_prefetch
        float4 f0 = s4[0], f1 = s4[1], f2 = s4[2], f3 = s4[3];
        v4u p0, p1;
        p0.x = (unsigned)f32_to_bf16_rne(f0.x) | ((unsigned)f32_to_bf16_rne(f0.y) << 16);
        p0.y = (unsigned)f32_to_bf16_rne(f0.z) | ((unsigned)f32_to_bf16_rne(f0.w) << 16);
        p0.z = (unsigned)f32_to_bf16_rne(f1.x) | ((unsigned)f32_to_bf16_rne(f1.y) << 16);
        p0.w = (unsigned)f32_to_bf16_rne(f1.z) | ((unsigned)f32_to_bf16_rne(f1.w) << 16);
        p1.x = (unsigned)f32_to_bf16_rne(f2.x) | ((unsigned)f32_to_bf16_rne(f2.y) << 16);
        p1.y = (unsigned)f32_to_bf16_rne(f2.z) | ((unsigned)f32_to_bf16_rne(f2.w) << 16);
        p1.z = (unsigned)f32_to_bf16_rne(f3.x) | ((unsigned)f32_to_bf16_rne(f3.y) << 16);
        p1.w = (unsigned)f32_to_bf16_rne(f3.z) | ((unsigned)f32_to_bf16_rne(f3.w) << 16);
        v4u* dst = (v4u*)&Als[buf][frow * LDS_PAD + fhalf * 16];
        dst[0] = p0;
        dst[1] = p1;
    };

    // B tile fill.
    auto fillB = [&](int buf, int kb) {
#if USE_TDM
        if (wave == 0)
            tdm_load_b_tile(W + (size_t)nBase * 4096 + kb,
                            lds_byte_addr(&Bls[buf][0]));
#else
        const v4u* src =
            (const v4u*)(W + (size_t)(nBase + frow) * 4096 + kb + fhalf * 16);
        v4u* dst = (v4u*)&Bls[buf][frow * LDS_PAD + fhalf * 16];
        dst[0] = src[0];
        dst[1] = src[1];
#endif
    };

    auto waitB = [&]() {
#if USE_TDM
        if (wave == 0) __builtin_amdgcn_s_wait_tensorcnt(0);
#endif
    };

    v8f acc[2][4];
    const v8f vzero = {0.f, 0.f, 0.f, 0.f, 0.f, 0.f, 0.f, 0.f};
    #pragma unroll
    for (int mi = 0; mi < 2; ++mi)
        #pragma unroll
        for (int ni = 0; ni < 4; ++ni) acc[mi][ni] = vzero;

    // ---- prologue: stage buffer 0 ----
    fillB(0, 0);
    fillA(0, 0);
    waitB();
    __syncthreads();

    int cur = 0;
    for (int kb = 0; kb < 4096; kb += TILE_K, cur ^= 1) {
        const int  nxt  = cur ^ 1;
        const bool more = (kb + TILE_K) < 4096;

        // ---- prefetch next tile into the other buffer ----
        if (more) {
            fillB(nxt, kb + TILE_K);
            fillA(nxt, kb + TILE_K);
        }

        // ---- fragments per ISA 16-bit A(16x32) / B(32x16) layouts ----
        FragBF afr[2], bfr[4];
        #pragma unroll
        for (int mi = 0; mi < 2; ++mi) {
            const unsigned short* ap =
                &Als[cur][(wm * 32 + mi * 16 + lm) * LDS_PAD + khalf * 8];
            afr[mi].q[0] = *(const v4u*)ap;          // K = khalf*8 .. +7
            afr[mi].q[1] = *(const v4u*)(ap + 16);   // K = khalf*8+16 .. +23
        }
        #pragma unroll
        for (int ni = 0; ni < 4; ++ni) {
            const unsigned short* bp =
                &Bls[cur][(wn * 64 + ni * 16 + lm) * LDS_PAD + khalf * 8];
            bfr[ni].q[0] = *(const v4u*)bp;          // K = khalf*8 .. +7
            bfr[ni].q[1] = *(const v4u*)(bp + 16);   // K = khalf*8+16 .. +23
        }

        #pragma unroll
        for (int mi = 0; mi < 2; ++mi)
            #pragma unroll
            for (int ni = 0; ni < 4; ++ni)
                acc[mi][ni] = __builtin_amdgcn_wmma_f32_16x16x32_bf16(
                    false, afr[mi].bf, false, bfr[ni].bf,
                    (short)0, acc[mi][ni], false, false);

        if (more) {
            waitB();          // TDM for buf[nxt] complete (wave 0)
            __syncthreads();  // A ds_stores visible; everyone done reading cur
        }
    }

    // ---- epilogue: C/D layout => lane<16: M=r, lane>=16: M=r+8; N=lane&15 ----
    const int moff = (lane >> 4) * 8;
    #pragma unroll
    for (int mi = 0; mi < 2; ++mi) {
        #pragma unroll
        for (int ni = 0; ni < 4; ++ni) {
            const int col = nBase + wn * 64 + ni * 16 + lm;
            const float b = bias[col];
            #pragma unroll
            for (int r = 0; r < 8; ++r) {
                const int row = mBase + wm * 32 + mi * 16 + moff + r;
                out[(size_t)row * 4096 + col] = acc[mi][ni][r] + b;
            }
        }
    }
}

// ---------------------------------------------------------------------------
// Launch. d_ws layout: [0, 128KB) g (f32, 64*512), [128KB, +32MB) W (bf16).
// Requires ws_size >= ~33.7 MB.
// ---------------------------------------------------------------------------
extern "C" void kernel_launch(void* const* d_in, const int* in_sizes, int n_in,
                              void* d_out, int out_size, void* d_ws, size_t ws_size,
                              hipStream_t stream) {
    const float* x         = (const float*)d_in[0];   // (8,2048,4096)
    const float* spec_re   = (const float*)d_in[1];   // (8,8,512)
    const float* spec_im   = (const float*)d_in[2];   // (8,8,512)
    const float* sign_flip = (const float*)d_in[3];   // (4096,)
    const float* bias      = (const float*)d_in[4];   // (4096,)
    float*       out       = (float*)d_out;           // (8,2048,4096)

    float*          g = (float*)d_ws;
    unsigned short* W = (unsigned short*)((char*)d_ws + (128u << 10));

    build_g_kernel<<<(64 * 512) / 256, 256, 0, stream>>>(spec_re, spec_im, g);
    build_w_kernel<<<(4096 * 4096) / 256, 256, 0, stream>>>(g, sign_flip, W);

    dim3 grid(4096 / TILE_N, 16384 / TILE_M);   // (32, 128)
    bcl_gemm_kernel<<<grid, 256, 0, stream>>>(x, W, bias, out);
}
